// PagedMixtralSparseMoeBlock_52201032515662
// MI455X (gfx1250) — matrix operand order repaired and checked by reading, not compiled, not run on hardware
//
#include <hip/hip_runtime.h>
#include <hip/hip_bf16.h>

// ---------------------------------------------------------------------------
// Mixtral-style sparse MoE block for gfx1250 (MI455X), bf16 WMMA path.
//   T=2048 tokens, H=1024, E=8, F=3584, top-2 routing (dense combine weights)
// Strategy:
//   - convert weights + activations to bf16 once into workspace (fits 192MB L2)
//   - compute transposed GEMMs (Ct = W * Xt) so A fragments are
//     contiguous b128 loads (A: weight rows K-contig; B: 16 bf16 loads at stride Tn from xT/hmidT)
//   - register-blocked macro-tiles, N(token)-dim widened to 64 to amortize the
//     weight stream: GEMM1 32x64 -> 16 wmma / 8 b128 + 64 bf16 loads per K-step,
//     GEMM2 32x64 -> 8 wmma / 4 b128 + 64 bf16 loads per K-step
//   - v_wmma_f32_16x16x32_bf16 with fp32 accumulation
// ---------------------------------------------------------------------------

typedef __attribute__((ext_vector_type(16))) __bf16 v16bf;
typedef __attribute__((ext_vector_type(8)))  __bf16 v8bf;
typedef __attribute__((ext_vector_type(8)))  float  v8f;

union V16 { v16bf v; v8bf h[2]; };

constexpr int Tn = 2048;   // tokens
constexpr int Hn = 1024;   // hidden
constexpr int En = 8;      // experts
constexpr int Fn = 3584;   // ffn dim

__device__ __forceinline__ __bf16 f2bf(float f) {
  unsigned u = __builtin_bit_cast(unsigned, f);
  u += 0x7FFFu + ((u >> 16) & 1u);               // round-to-nearest-even
  unsigned short s = (unsigned short)(u >> 16);
  return __builtin_bit_cast(__bf16, s);
}

__device__ __forceinline__ v8f wmma_bf16(v16bf a, v16bf b, v8f c) {
  // (neg_a, A, neg_b, B, c_mod, C, reuse_a, reuse_b)
  return __builtin_amdgcn_wmma_f32_16x16x32_bf16(false, a, false, b,
                                                 (short)0, c, false, false);
}

// A fragment: per-lane 8 bf16 at +0 and 8 bf16 at +16 (K-interleaved halves)
__device__ __forceinline__ V16 load_fragA(const __bf16* p) {
  V16 r;
  r.h[0] = *(const v8bf*)(p);
  r.h[1] = *(const v8bf*)(p + 16);
  return r;
}
// B fragment: per-lane column n, K rows +0..7 and +16..23 (stride Tn) of xT/hmidT
__device__ __forceinline__ V16 load_fragB(const __bf16* p) {
  V16 r;
#pragma unroll
  for (int s = 0; s < 8; ++s) {
    r.v[s] = p[(size_t)s * Tn];
    r.v[8 + s] = p[(size_t)(16 + s) * Tn];
  }
  return r;
}

// ------------------------------- utility kernels ---------------------------

__global__ __launch_bounds__(256) void zero_kernel(float4* p, int n4) {
  int i = blockIdx.x * 256 + threadIdx.x;
  if (i < n4) p[i] = make_float4(0.f, 0.f, 0.f, 0.f);
}

__global__ __launch_bounds__(256) void cvt_kernel(const float* __restrict__ src,
                                                  __bf16* __restrict__ dst, int n4) {
  int i = blockIdx.x * 256 + threadIdx.x;
  if (i < n4) {
    float4 v = ((const float4*)src)[i];
    ushort4 o;
    o.x = __builtin_bit_cast(unsigned short, f2bf(v.x));
    o.y = __builtin_bit_cast(unsigned short, f2bf(v.y));
    o.z = __builtin_bit_cast(unsigned short, f2bf(v.z));
    o.w = __builtin_bit_cast(unsigned short, f2bf(v.w));
    ((ushort4*)dst)[i] = o;
  }
}

// x[T,H] fp32 -> xT[H,T] bf16, LDS 32x33 tile transpose
__global__ __launch_bounds__(256) void transpose_cvt_kernel(const float* __restrict__ x,
                                                            __bf16* __restrict__ xT) {
  __shared__ __bf16 tile[32][33];
  int tx = threadIdx.x & 31;   // 0..31
  int ty = threadIdx.x >> 5;   // 0..7
  int h  = blockIdx.x * 32 + tx;
  int t0 = blockIdx.y * 32;
#pragma unroll
  for (int j = 0; j < 32; j += 8)
    tile[ty + j][tx] = f2bf(x[(size_t)(t0 + ty + j) * Hn + h]);
  __syncthreads();
#pragma unroll
  for (int j = 0; j < 32; j += 8)
    xT[(size_t)(blockIdx.x * 32 + ty + j) * Tn + t0 + tx] = tile[tx][ty + j];
}

// ------------------------------- router ------------------------------------
// one wave per token: logits = x @ gate_w.T, softmax, top-2, renorm -> combine
__global__ __launch_bounds__(256) void router_kernel(const float* __restrict__ x,
                                                     const float* __restrict__ gw,
                                                     float* __restrict__ logits_out,
                                                     float* __restrict__ combine) {
  int wave = threadIdx.x >> 5;
  int lane = threadIdx.x & 31;
  int t = blockIdx.x * 8 + wave;
  const float* xr = x + (size_t)t * Hn;
  float lg[En];
#pragma unroll
  for (int e = 0; e < En; ++e) {
    const float* g = gw + (size_t)e * Hn;
    float s = 0.f;
    for (int h = lane; h < Hn; h += 32) s = fmaf(xr[h], g[h], s);
#pragma unroll
    for (int m = 16; m >= 1; m >>= 1) s += __shfl_xor(s, m, 32);
    lg[e] = s;                               // every lane holds the total
  }
  if (lane < En) logits_out[(size_t)t * En + lane] = lg[lane];
  if (lane == 0) {
    float mx = lg[0];
#pragma unroll
    for (int e = 1; e < En; ++e) mx = fmaxf(mx, lg[e]);
    float p[En]; float sum = 0.f;
#pragma unroll
    for (int e = 0; e < En; ++e) { p[e] = __expf(lg[e] - mx); sum += p[e]; }
#pragma unroll
    for (int e = 0; e < En; ++e) p[e] /= sum;
    int i1 = 0;
#pragma unroll
    for (int e = 1; e < En; ++e) if (p[e] > p[i1]) i1 = e;
    int i2 = (i1 == 0) ? 1 : 0;
#pragma unroll
    for (int e = 0; e < En; ++e) if (e != i1 && p[e] > p[i2]) i2 = e;
    float pairsum = p[i1] + p[i2];
#pragma unroll
    for (int e = 0; e < En; ++e) combine[(size_t)t * En + e] = 0.f;
    combine[(size_t)t * En + i1] = p[i1] / pairsum;
    combine[(size_t)t * En + i2] = p[i2] / pairsum;
  }
}

// ------------------------------- GEMM1 -------------------------------------
// hmidT[F,T] = silu(W1 * xT) * (W3 * xT), one 32(F) x 64(T) macro-tile / wave.
// Per K-step: 4 A-frags (w1 lo/hi, w3 lo/hi) + 4 B-frags -> 16 wmma.
__global__ __launch_bounds__(256) void gemm1_kernel(const __bf16* __restrict__ w1b,
                                                    const __bf16* __restrict__ w3b,
                                                    const __bf16* __restrict__ xT,
                                                    const float* __restrict__ combine,
                                                    int e,
                                                    __bf16* __restrict__ hT) {
  int wid  = blockIdx.x * 8 + (threadIdx.x >> 5);
  int lane = threadIdx.x & 31;
  int tt = wid & 31;           // T/64 = 32 tiles; T fastest -> weights stream once
  int ft = wid >> 5;           // 0..111
  int t0 = tt * 64, f0 = ft * 32;
  int half = lane >> 4;        // 0/1
  int lr   = lane & 15;

  // whole-macro-tile skip if no token in [t0, t0+64) routes to expert e
  float cw0 = combine[(size_t)(t0 + lane) * En + e];
  float cw1 = combine[(size_t)(t0 + 32 + lane) * En + e];
  if (__ballot((cw0 != 0.f) || (cw1 != 0.f)) == 0ull) return;

  const __bf16* wa1 = w1b + (size_t)(f0 + lr) * Hn + half * 8;
  const __bf16* wa3 = w3b + (size_t)(f0 + lr) * Hn + half * 8;
  const __bf16* xb  = xT + (size_t)(8 * half) * Tn + t0 + lr;

  v8f c1[2][4] = {}; v8f c3[2][4] = {};
  for (int k0 = 0; k0 < Hn; k0 += 32) {
    V16 a10 = load_fragA(wa1 + k0);
    V16 a11 = load_fragA(wa1 + (size_t)16 * Hn + k0);
    V16 a30 = load_fragA(wa3 + k0);
    V16 a31 = load_fragA(wa3 + (size_t)16 * Hn + k0);
    const __bf16* xp = xb + (size_t)k0 * Tn;
#pragma unroll
    for (int ni = 0; ni < 4; ++ni) {
      V16 b = load_fragB(xp + ni * 16);
      c1[0][ni] = wmma_bf16(a10.v, b.v, c1[0][ni]);
      c1[1][ni] = wmma_bf16(a11.v, b.v, c1[1][ni]);
      c3[0][ni] = wmma_bf16(a30.v, b.v, c3[0][ni]);
      c3[1][ni] = wmma_bf16(a31.v, b.v, c3[1][ni]);
    }
  }
  // D layout: vgpr r, lanes 0-15 -> M=r, lanes 16-31 -> M=8+r; N=lane&15
#pragma unroll
  for (int mi = 0; mi < 2; ++mi)
#pragma unroll
    for (int ni = 0; ni < 4; ++ni)
#pragma unroll
      for (int r = 0; r < 8; ++r) {
        float v1 = c1[mi][ni][r], v3 = c3[mi][ni][r];
        float act = (v1 / (1.f + __expf(-v1))) * v3;   // silu(v1)*v3
        int f = f0 + mi * 16 + half * 8 + r;
        hT[(size_t)f * Tn + t0 + ni * 16 + lr] = f2bf(act);
      }
}

// ------------------------------- GEMM2 -------------------------------------
// outT tile = W2 * hmidT; 32(H) x 64(T) macro-tile per wave; scale columns by
// combine[t,e] and accumulate fp32 into out[T,H].
__global__ __launch_bounds__(256) void gemm2_kernel(const __bf16* __restrict__ w2b,
                                                    const __bf16* __restrict__ hT,
                                                    const float* __restrict__ combine,
                                                    int e,
                                                    float* __restrict__ out) {
  int wid  = blockIdx.x * 8 + (threadIdx.x >> 5);
  int lane = threadIdx.x & 31;
  int tt = wid & 31;           // T/64 = 32; T fastest: hmidT replays from L2
  int ht = wid >> 5;           // 0..31
  int t0 = tt * 64, h0 = ht * 32;
  int half = lane >> 4;
  int lr   = lane & 15;

  float cw0 = combine[(size_t)(t0 + lane) * En + e];
  float cw1 = combine[(size_t)(t0 + 32 + lane) * En + e];
  if (__ballot((cw0 != 0.f) || (cw1 != 0.f)) == 0ull) return;  // match gemm1

  const __bf16* wa = w2b + (size_t)(h0 + lr) * Fn + half * 8;
  const __bf16* hb = hT + (size_t)(8 * half) * Tn + t0 + lr;

  v8f c[2][4] = {};
  for (int k0 = 0; k0 < Fn; k0 += 32) {
    V16 a0 = load_fragA(wa + k0);
    V16 a1 = load_fragA(wa + (size_t)16 * Fn + k0);
    const __bf16* hp = hb + (size_t)k0 * Tn;
#pragma unroll
    for (int ni = 0; ni < 4; ++ni) {
      V16 b = load_fragB(hp + ni * 16);
      c[0][ni] = wmma_bf16(a0.v, b.v, c[0][ni]);
      c[1][ni] = wmma_bf16(a1.v, b.v, c[1][ni]);
    }
  }
#pragma unroll
  for (int ni = 0; ni < 4; ++ni) {
    int t = t0 + ni * 16 + lr;
    float cw = combine[(size_t)t * En + e];
#pragma unroll
    for (int mi = 0; mi < 2; ++mi) {
      float* p = out + (size_t)t * Hn + h0 + mi * 16 + half * 8;
      float4 o0 = *(float4*)p;
      float4 o1 = *(float4*)(p + 4);
      o0.x += cw * c[mi][ni][0]; o0.y += cw * c[mi][ni][1];
      o0.z += cw * c[mi][ni][2]; o0.w += cw * c[mi][ni][3];
      o1.x += cw * c[mi][ni][4]; o1.y += cw * c[mi][ni][5];
      o1.z += cw * c[mi][ni][6]; o1.w += cw * c[mi][ni][7];
      *(float4*)p       = o0;
      *(float4*)(p + 4) = o1;
    }
  }
}

// ------------------------------- launcher ----------------------------------

extern "C" void kernel_launch(void* const* d_in, const int* in_sizes, int n_in,
                              void* d_out, int out_size, void* d_ws, size_t ws_size,
                              hipStream_t stream) {
  (void)in_sizes; (void)n_in; (void)out_size; (void)ws_size;
  const float* x  = (const float*)d_in[0];   // [T,H]
  const float* gw = (const float*)d_in[1];   // [E,H]
  const float* w1 = (const float*)d_in[2];   // [E,F,H]
  const float* w2 = (const float*)d_in[3];   // [E,H,F]
  const float* w3 = (const float*)d_in[4];   // [E,F,H]
  float* out    = (float*)d_out;             // [T,H]
  float* logits = out + (size_t)Tn * Hn;     // [T,E] (second tuple output)

  char* p = (char*)d_ws;
  auto alloc = [&](size_t bytes) {
    char* r = p;
    p += (bytes + 255) & ~(size_t)255;
    return r;
  };
  __bf16* xT   = (__bf16*)alloc((size_t)Hn * Tn * 2);       //  4 MB
  float*  comb = (float*) alloc((size_t)Tn * En * 4);       // 64 KB
  __bf16* w1b  = (__bf16*)alloc((size_t)En * Fn * Hn * 2);  // 58.7 MB
  __bf16* w3b  = (__bf16*)alloc((size_t)En * Fn * Hn * 2);  // 58.7 MB
  __bf16* w2b  = (__bf16*)alloc((size_t)En * Hn * Fn * 2);  // 58.7 MB
  __bf16* hT   = (__bf16*)alloc((size_t)Fn * Tn * 2);       // 14.7 MB

  const int wn4 = En * Fn * Hn / 4;  // fp32->bf16 quads per weight tensor

  zero_kernel<<<(Tn * Hn / 4) / 256, 256, 0, stream>>>((float4*)out, Tn * Hn / 4);
  transpose_cvt_kernel<<<dim3(Hn / 32, Tn / 32), 256, 0, stream>>>(x, xT);
  cvt_kernel<<<wn4 / 256, 256, 0, stream>>>(w1, w1b, wn4);
  cvt_kernel<<<wn4 / 256, 256, 0, stream>>>(w3, w3b, wn4);
  cvt_kernel<<<wn4 / 256, 256, 0, stream>>>(w2, w2b, wn4);
  router_kernel<<<Tn / 8, 256, 0, stream>>>(x, gw, logits, comb);

  const int g1_blocks = (Fn / 32) * (Tn / 64) / 8;  // 448
  const int g2_blocks = (Hn / 32) * (Tn / 64) / 8;  // 128
  for (int e = 0; e < En; ++e) {
    gemm1_kernel<<<g1_blocks, 256, 0, stream>>>(w1b + (size_t)e * Fn * Hn,
                                                w3b + (size_t)e * Fn * Hn,
                                                xT, comb, e, hT);
    gemm2_kernel<<<g2_blocks, 256, 0, stream>>>(w2b + (size_t)e * Hn * Fn,
                                                hT, comb, e, out);
  }
}
